// Informer_18966575579402
// MI455X (gfx1250) — matrix operand order repaired and checked by reading, not compiled, not run on hardware
//
#include <hip/hip_runtime.h>
#include <math.h>
#include <stdint.h>

// ---------------------------------------------------------------------------
// Informer forward for MI455X (gfx1250, wave32, WMMA).
// Heavy GEMMs -> v_wmma_f32_16x16x32_bf16 (bf16 in, f32 accum),
//               128x128 block tile, 8 waves, async global->LDS pipeline
//               (GLOBAL_LOAD_ASYNC_TO_LDS_B128 + s_wait_asynccnt).
// Sparse attention (u=35) -> small VALU kernels.
// ---------------------------------------------------------------------------

typedef __attribute__((ext_vector_type(16))) __bf16 v16bf;
typedef __attribute__((ext_vector_type(8)))  float  v8f;
typedef __attribute__((ext_vector_type(4)))  int    v4i;

#define T_LEN   1024
#define N_B     8
#define NX_IN   16
#define DM      512
#define DFF     2048
#define NH      8
#define DH      64
#define NT      8192          // N_B * T_LEN
#define U_TOP   35            // min(5*ceil(ln 1024), 1024)
#define LDST    40            // LDS tile row stride in halves (80B: 16B-aligned,
                              // bank-conflict-free: r*20 mod 64 distinct)

#if defined(__has_builtin)
#  if __has_builtin(__builtin_amdgcn_global_load_async_to_lds_b128)
#    define HAS_ASYNC_LDS 1
#  endif
#  if __has_builtin(__builtin_amdgcn_s_wait_asynccnt)
#    define HAS_WAIT_ASYNC 1
#  endif
#endif

#define AS1 __attribute__((address_space(1)))
#define AS3 __attribute__((address_space(3)))

// ------------------------- helpers -----------------------------------------

__host__ __device__ inline void threefry2x32(unsigned k0, unsigned k1,
                                             unsigned c0, unsigned c1,
                                             unsigned* o0, unsigned* o1) {
  unsigned ks[3] = {k0, k1, k0 ^ k1 ^ 0x1BD11BDAu};
  unsigned x0 = c0 + k0, x1 = c1 + k1;
  const int rot[2][4] = {{13, 15, 26, 6}, {17, 29, 16, 24}};
  for (int i = 0; i < 5; ++i) {
    const int* r = rot[i & 1];
    for (int j = 0; j < 4; ++j) {
      x0 += x1;
      x1 = (x1 << r[j]) | (x1 >> (32 - r[j]));
      x1 ^= x0;
    }
    x0 += ks[(i + 1) % 3];
    x1 += ks[(i + 2) % 3] + (unsigned)(i + 1);
  }
  *o0 = x0; *o1 = x1;
}

__device__ inline unsigned short f2bf(float f) {   // round-to-nearest-even
  unsigned u = __float_as_uint(f);
  unsigned r = u + 0x7FFFu + ((u >> 16) & 1u);
  return (unsigned short)(r >> 16);
}

__device__ inline void wait_async() {
#if defined(HAS_ASYNC_LDS)
# if defined(HAS_WAIT_ASYNC)
  __builtin_amdgcn_s_wait_asynccnt(0);
# else
  asm volatile("s_wait_asynccnt 0" ::: "memory");
# endif
#endif
}

// Stage one 128x32-half tile row-chunk per thread (2 x 16B).
// Async path: GLOBAL_LOAD_ASYNC_TO_LDS_B128 (ASYNCcnt); else reg round-trip.
__device__ inline void stage_tile(const unsigned short* __restrict__ base,
                                  unsigned short* sh, int rowBase, int K,
                                  int k0, int tid) {
  int row = tid >> 1, c0 = (tid & 1) * 16;
  const unsigned short* g = base + (size_t)(rowBase + row) * K + k0 + c0;
  unsigned short* s = sh + row * LDST + c0;
#if defined(HAS_ASYNC_LDS)
  __builtin_amdgcn_global_load_async_to_lds_b128(
      (AS1 v4i*)(uintptr_t)g, (AS3 v4i*)(uintptr_t)s, 0, 0);
  __builtin_amdgcn_global_load_async_to_lds_b128(
      (AS1 v4i*)(uintptr_t)(g + 8), (AS3 v4i*)(uintptr_t)(s + 8), 0, 0);
#else
  const unsigned* gu = (const unsigned*)g;
  unsigned* su = (unsigned*)s;
  #pragma unroll
  for (int i = 0; i < 8; ++i) su[i] = gu[i];
#endif
}

// ------------------------- kernels -----------------------------------------

// fp32 -> bf16 conversion
__global__ void k_cvt(const float* __restrict__ src,
                      unsigned short* __restrict__ dst, int n) {
  int g = blockIdx.x * blockDim.x + threadIdx.x;
  if (g < n) dst[g] = f2bf(src[g]);
}

// fp32 (K x N, row-major) -> bf16 transposed (N x K): weights for GEMM B-side
__global__ void k_cvtT(const float* __restrict__ src,
                       unsigned short* __restrict__ dst, int K, int N) {
  int g = blockIdx.x * blockDim.x + threadIdx.x;
  if (g >= K * N) return;
  int k = g / N, n = g - k * N;              // reads coalesced along n
  dst[(size_t)n * K + k] = f2bf(src[g]);
}

// token conv (circular, kernel 3) + sinusoidal positional embedding
__global__ void k_embed(const float* __restrict__ x,
                        const float* __restrict__ conv,
                        float* __restrict__ h) {
  int nt = blockIdx.x;            // row = n*T + t
  int n  = nt >> 10;
  int t  = nt & (T_LEN - 1);
  __shared__ float xs[48];
  int tid = threadIdx.x;
  if (tid < 48) {
    int w = tid >> 4, ci = tid & 15;
    int tt = (t + w - 1 + T_LEN) & (T_LEN - 1);   // circular pad
    xs[tid] = x[(size_t)(tt * N_B + n) * NX_IN + ci];
  }
  __syncthreads();
  for (int c = tid; c < DM; c += blockDim.x) {
    float acc = 0.f;
    #pragma unroll
    for (int w = 0; w < 3; ++w)
      #pragma unroll
      for (int ci = 0; ci < NX_IN; ++ci)
        acc += xs[w * 16 + ci] * conv[(size_t)(w * 16 + ci) * DM + c];
    int   i   = c >> 1;
    float div = expf((float)(2 * i) * (-9.210340371976184f / (float)DM));
    float ang = (float)t * div;
    float pe  = (c & 1) ? cosf(ang) : sinf(ang);
    h[(size_t)nt * DM + c] = acc + pe;
  }
}

// Tiled bf16 WMMA GEMM: C[M,N] = A[M,K] * Bt[N,K]^T + bias, optional GELU,
// optional bf16 output.  M,N multiples of 128, K multiple of 32.
// 256 threads = 8 waves; wave grid 4(M) x 2(N); 32x64 per wave (2x4 WMMA acc).
__global__ __launch_bounds__(256)
void k_gemm(const unsigned short* __restrict__ A,
            const unsigned short* __restrict__ Bt,
            const float* __restrict__ bias,
            float* __restrict__ Cf, unsigned short* __restrict__ Cb,
            int M, int N, int K, int act, int outBf) {
  __shared__ unsigned short As[128 * LDST];
  __shared__ unsigned short Bs[128 * LDST];
  const int tid     = threadIdx.x;
  const int nTilesN = N >> 7;
  const int bm      = (blockIdx.x / nTilesN) << 7;
  const int bn      = (blockIdx.x % nTilesN) << 7;
  const int wave = tid >> 5, lane = tid & 31;
  const int half = lane >> 4, r = lane & 15;
  const int wm = (wave >> 1) << 5;      // 0,32,64,96
  const int wn = (wave & 1) << 6;       // 0,64

  const v8f vzero = {0.f, 0.f, 0.f, 0.f, 0.f, 0.f, 0.f, 0.f};
  v8f acc[2][4];
  #pragma unroll
  for (int i = 0; i < 2; ++i)
    #pragma unroll
    for (int j = 0; j < 4; ++j) acc[i][j] = vzero;

  union Frag { v16bf v; unsigned u[8]; };

  // prologue: stage first K-tile
  stage_tile(A,  As, bm, K, 0, tid);
  stage_tile(Bt, Bs, bn, K, 0, tid);
  wait_async();
  __syncthreads();

  for (int k0 = 0; k0 < K; k0 += 32) {
    // fragment loads (current tile)
    Frag a[2], b[4];
    #pragma unroll
    for (int i = 0; i < 2; ++i) {
      // A frag: ISA 16-bit A layout -> lane-half selects K+8; dwords = K pairs
      const unsigned short* ar = As + (wm + i * 16 + r) * LDST + half * 8;
      #pragma unroll
      for (int v = 0; v < 8; ++v) {
        int ka = ((2 * v) & 7) + ((v >> 2) << 4);
        a[i].u[v] = *(const unsigned*)(ar + ka);
      }
    }
    #pragma unroll
    for (int j = 0; j < 4; ++j) {
      // B frag: same K order as A (lane-half selects K+8; dwords = K pairs)
      const unsigned short* br = Bs + (wn + j * 16 + r) * LDST + half * 8;
      #pragma unroll
      for (int v = 0; v < 8; ++v)
        b[j].u[v] = *(const unsigned*)(br + ((2 * v) & 7) + ((v >> 2) << 4));
    }
    __syncthreads();                  // all waves done reading LDS tiles

    // software pipeline: kick next tile's async fills, then do the math
    if (k0 + 32 < K) {
      stage_tile(A,  As, bm, K, k0 + 32, tid);
      stage_tile(Bt, Bs, bn, K, k0 + 32, tid);
    }

    #pragma unroll
    for (int i = 0; i < 2; ++i)
      #pragma unroll
      for (int j = 0; j < 4; ++j)
        acc[i][j] = __builtin_amdgcn_wmma_f32_16x16x32_bf16(
            false, a[i].v, false, b[j].v, (short)0, acc[i][j], false, false);

    wait_async();
    __syncthreads();
  }

  // epilogue: C layout -> M = e + 8*half, N = r within each 16x16 tile
  #pragma unroll
  for (int i = 0; i < 2; ++i) {
    #pragma unroll
    for (int j = 0; j < 4; ++j) {
      int   col = bn + wn + j * 16 + r;
      float bv  = bias ? bias[col] : 0.f;
      #pragma unroll
      for (int e = 0; e < 8; ++e) {
        int   row = bm + wm + i * 16 + e + half * 8;
        float v   = acc[i][j][e] + bv;
        if (act == 1) v = 0.5f * v * (1.0f + erff(v * 0.70710678118654752f));
        size_t o = (size_t)row * N + col;
        if (outBf) Cb[o] = f2bf(v);
        else       Cf[o] = v;
      }
    }
  }
}

// threefry sampling: idx[l*u+s] = bits & 1023 (span 1024 is a power of two)
__global__ void k_sample(int* __restrict__ sidx, unsigned k0, unsigned k1) {
  int j = blockIdx.x * blockDim.x + threadIdx.x;
  const int halfn = (T_LEN * U_TOP) / 2;          // 17920
  if (j >= halfn) return;
  unsigned o0, o1;
  threefry2x32(k0, k1, (unsigned)j, (unsigned)(j + halfn), &o0, &o1);
  sidx[j]         = (int)(o0 & (T_LEN - 1));
  sidx[j + halfn] = (int)(o1 & (T_LEN - 1));
}

// sparsity measurement M = max_s(qk_s) - sum_s(qk_s)/L
__global__ void k_probM(const float* __restrict__ q,
                        const float* __restrict__ k,
                        const int* __restrict__ sidx,
                        float* __restrict__ Msc) {
  int g = blockIdx.x * blockDim.x + threadIdx.x;
  if (g >= N_B * NH * T_LEN) return;
  int l = g & (T_LEN - 1), bh = g >> 10, b = bh >> 3, hh = bh & 7;
  const float* qr = q + ((size_t)(b * T_LEN + l) * DM + hh * DH);
  float qv[DH];
  #pragma unroll
  for (int d = 0; d < DH; ++d) qv[d] = qr[d];
  float mx = -3.4e38f, sm = 0.f;
  for (int s = 0; s < U_TOP; ++s) {
    int ls = sidx[l * U_TOP + s];
    const float* kr = k + ((size_t)(b * T_LEN + ls) * DM + hh * DH);
    float d0 = 0.f;
    #pragma unroll
    for (int d = 0; d < DH; ++d) d0 += qv[d] * kr[d];
    mx = fmaxf(mx, d0);
    sm += d0;
  }
  Msc[(size_t)bh * T_LEN + l] = mx - sm * (1.0f / (float)T_LEN);
}

// iterative top-k (tie -> smallest index, matches lax.top_k ordering)
__global__ void k_topk(const float* __restrict__ Msc, int* __restrict__ top) {
  int bh = blockIdx.x;
  __shared__ float vals[T_LEN];
  __shared__ float rv[256];
  __shared__ int   ri[256];
  int tid = threadIdx.x;
  for (int i = tid; i < T_LEN; i += 256) vals[i] = Msc[(size_t)bh * T_LEN + i];
  __syncthreads();
  for (int j = 0; j < U_TOP; ++j) {
    float bv = -3.4e38f; int bi = 0x7fffffff;
    for (int i = tid; i < T_LEN; i += 256) {
      float v = vals[i];
      if (v > bv || (v == bv && i < bi)) { bv = v; bi = i; }
    }
    rv[tid] = bv; ri[tid] = bi;
    __syncthreads();
    for (int s = 128; s > 0; s >>= 1) {
      if (tid < s) {
        if (rv[tid + s] > rv[tid] ||
            (rv[tid + s] == rv[tid] && ri[tid + s] < ri[tid])) {
          rv[tid] = rv[tid + s]; ri[tid] = ri[tid + s];
        }
      }
      __syncthreads();
    }
    if (tid == 0) { top[bh * U_TOP + j] = ri[0]; vals[ri[0]] = -3.4e38f; }
    __syncthreads();
  }
}

// mean of V over L, per (b,h,d)
__global__ void k_vmean(const float* __restrict__ v, float* __restrict__ vm) {
  int g = blockIdx.x * blockDim.x + threadIdx.x;
  if (g >= N_B * NH * DH) return;
  int d = g & 63, bh = g >> 6, b = bh >> 3, hh = bh & 7;
  float s = 0.f;
  for (int l = 0; l < T_LEN; ++l)
    s += v[(size_t)(b * T_LEN + l) * DM + hh * DH + d];
  vm[g] = s * (1.0f / (float)T_LEN);
}

// fill ctx (stored bf16 in (N,T,DM) layout) with broadcast v-mean
__global__ void k_ctxfill(const float* __restrict__ vm,
                          unsigned short* __restrict__ ctxb) {
  size_t g = (size_t)blockIdx.x * blockDim.x + threadIdx.x;
  if (g >= (size_t)NT * DM) return;
  int c = (int)(g & (DM - 1));
  int row = (int)(g >> 9);
  int b = row >> 10;
  int hh = c >> 6, d = c & 63;
  ctxb[g] = f2bf(vm[(b * NH + hh) * DH + d]);
}

// full attention for the selected (top) queries, scatter into ctx
__global__ void k_attn(const float* __restrict__ q,
                       const float* __restrict__ k,
                       const float* __restrict__ v,
                       const int* __restrict__ top,
                       unsigned short* __restrict__ ctxb) {
  int blk = blockIdx.x;                 // 64 * 35
  int bh = blk / U_TOP, j = blk % U_TOP;
  int b = bh >> 3, hh = bh & 7;
  __shared__ float sc[T_LEN];
  __shared__ float qs[DH];
  __shared__ float red[256];
  int tid  = threadIdx.x;
  int qrow = top[bh * U_TOP + j];
  if (tid < DH) qs[tid] = q[(size_t)(b * T_LEN + qrow) * DM + hh * DH + tid];
  __syncthreads();
  for (int l = tid; l < T_LEN; l += 256) {
    const float* kr = k + ((size_t)(b * T_LEN + l) * DM + hh * DH);
    float d0 = 0.f;
    #pragma unroll
    for (int d = 0; d < DH; ++d) d0 += qs[d] * kr[d];
    sc[l] = d0 * 0.125f;                 // 1/sqrt(64)
  }
  __syncthreads();
  float m = -3.4e38f;
  for (int l = tid; l < T_LEN; l += 256) m = fmaxf(m, sc[l]);
  red[tid] = m; __syncthreads();
  for (int s = 128; s > 0; s >>= 1) {
    if (tid < s) red[tid] = fmaxf(red[tid], red[tid + s]);
    __syncthreads();
  }
  m = red[0];
  __syncthreads();
  float ssum = 0.f;
  for (int l = tid; l < T_LEN; l += 256) {
    float e = expf(sc[l] - m);
    sc[l] = e; ssum += e;
  }
  red[tid] = ssum; __syncthreads();
  for (int s = 128; s > 0; s >>= 1) {
    if (tid < s) red[tid] += red[tid + s];
    __syncthreads();
  }
  float inv = 1.0f / red[0];
  __syncthreads();
  if (tid < DH) {
    int d = tid; float a = 0.f;
    for (int l = 0; l < T_LEN; ++l)
      a += sc[l] * v[(size_t)(b * T_LEN + l) * DM + hh * DH + d];
    ctxb[(size_t)(b * T_LEN + qrow) * DM + hh * DH + d] = f2bf(a * inv);
  }
}

// LayerNorm over last dim (512); optional residual add; may run in place
__global__ void k_ln(const float* __restrict__ xin,
                     const float* __restrict__ res,
                     const float* __restrict__ g,
                     const float* __restrict__ bta,
                     float* __restrict__ out) {
  int row = blockIdx.x;
  __shared__ float red[256];
  __shared__ float xs[DM];
  int tid = threadIdx.x;
  float s = 0.f;
  for (int c = tid; c < DM; c += 256) {
    float v = xin[(size_t)row * DM + c];
    if (res) v += res[(size_t)row * DM + c];
    xs[c] = v; s += v;
  }
  red[tid] = s; __syncthreads();
  for (int st = 128; st > 0; st >>= 1) {
    if (tid < st) red[tid] += red[tid + st];
    __syncthreads();
  }
  float mean = red[0] * (1.0f / (float)DM);
  __syncthreads();
  float vs = 0.f;
  for (int c = tid; c < DM; c += 256) { float d = xs[c] - mean; vs += d * d; }
  red[tid] = vs; __syncthreads();
  for (int st = 128; st > 0; st >>= 1) {
    if (tid < st) red[tid] += red[tid + st];
    __syncthreads();
  }
  float inv = 1.0f / sqrtf(red[0] * (1.0f / (float)DM) + 1e-5f);
  for (int c = tid; c < DM; c += 256)
    out[(size_t)row * DM + c] = (xs[c] - mean) * inv * g[c] + bta[c];
}

// final projection (512 -> 1) with (N,T)->(T,N) transpose
__global__ void k_proj(const float* __restrict__ h,
                       const float* __restrict__ Wp,
                       const float* __restrict__ bp,
                       float* __restrict__ out) {
  int g = blockIdx.x * blockDim.x + threadIdx.x;
  if (g >= NT) return;
  int t = g >> 3, n = g & 7;
  const float* hr = h + (size_t)(n * T_LEN + t) * DM;
  float s = 0.f;
  for (int c = 0; c < DM; ++c) s += hr[c] * Wp[c];
  out[g] = s + bp[0];                    // out[(t*N + n)]
}

// ------------------------- host orchestration -------------------------------

extern "C" void kernel_launch(void* const* d_in, const int* in_sizes, int n_in,
                              void* d_out, int out_size, void* d_ws,
                              size_t ws_size, hipStream_t stream) {
  (void)in_sizes; (void)out_size; (void)ws_size;
  if (n_in < 7 + 2 * 16) return;

  auto F = [&](int i) { return (const float*)d_in[i]; };
  const float* x    = F(0);
  const float* conv = F(1);
  // F(2) = time_w : multiplied by zeros in the reference -> skipped
  const float* ln_g = F(3);
  const float* ln_b = F(4);
  const float* Wp   = F(5);
  const float* bp   = F(6);

  char* ws = (char*)d_ws;
  const size_t MB = 1u << 20;
  float*          h    = (float*)(ws);                     // 16 MB
  float*          t0   = (float*)(ws + 16 * MB);           // 16 MB
  float*          qf   = (float*)(ws + 32 * MB);           // 16 MB
  float*          kf   = (float*)(ws + 48 * MB);           // 16 MB
  float*          vf   = (float*)(ws + 64 * MB);           // 16 MB
  unsigned short* abf  = (unsigned short*)(ws + 80 * MB);  // 32 MB (bf16 act)
  unsigned short* bbf  = (unsigned short*)(ws + 112 * MB); // 32 MB (bf16 FFN)
  unsigned short* wbf  = (unsigned short*)(ws + 144 * MB); // 2 MB  (bf16 wts^T)
  float*          Msc  = (float*)(ws + 146 * MB);          // 256 KB
  int*            sidx = (int*)(ws + 146 * MB + 512 * 1024); // 140 KB
  int*            topb = (int*)(ws + 147 * MB);            // 9 KB
  float*          vmn  = (float*)(ws + 147 * MB + 64 * 1024); // 16 KB

  auto cvt = [&](const float* s, unsigned short* d, int n) {
    k_cvt<<<(n + 255) / 256, 256, 0, stream>>>(s, d, n);
  };
  auto gemm = [&](const unsigned short* A, const float* W, int wN, int wK,
                  const float* bias, float* Cf, unsigned short* Cb,
                  int act, int outBf) {
    int ne = wK * wN;
    k_cvtT<<<(ne + 255) / 256, 256, 0, stream>>>(W, wbf, wK, wN);
    int grid = (NT / 128) * (wN / 128);
    k_gemm<<<grid, 256, 0, stream>>>(A, wbf, bias, Cf, Cb, NT, wN, wK, act, outBf);
  };

  // --- embedding ---
  k_embed<<<NT, 256, 0, stream>>>(x, conv, h);

  // --- transformer layers ---
  for (int L = 0; L < 2; ++L) {
    int p = 7 + 16 * L;
    const float *Wq = F(p + 0), *bq = F(p + 1), *Wk = F(p + 2), *bk = F(p + 3);
    const float *Wv = F(p + 4), *bv = F(p + 5), *Wo = F(p + 6), *bo = F(p + 7);
    const float *W1 = F(p + 8), *b1 = F(p + 9), *W2 = F(p + 10), *b2 = F(p + 11);
    const float *g1 = F(p + 12), *be1 = F(p + 13), *g2 = F(p + 14), *be2 = F(p + 15);

    // Q, K, V projections (WMMA)
    cvt(h, abf, NT * DM);
    gemm(abf, Wq, DM, DM, bq, qf, nullptr, 0, 0);
    gemm(abf, Wk, DM, DM, bk, kf, nullptr, 0, 0);
    gemm(abf, Wv, DM, DM, bv, vf, nullptr, 0, 0);

    // ProbSparse attention
    unsigned fk0, fk1;                    // fold_in(key(42), L)
    threefry2x32(0u, 42u, 0u, (unsigned)L, &fk0, &fk1);
    k_sample<<<((T_LEN * U_TOP / 2) + 255) / 256, 256, 0, stream>>>(sidx, fk0, fk1);
    k_probM<<<(N_B * NH * T_LEN) / 256, 256, 0, stream>>>(qf, kf, sidx, Msc);
    k_topk<<<N_B * NH, 256, 0, stream>>>(Msc, topb);
    k_vmean<<<(N_B * NH * DH + 255) / 256, 256, 0, stream>>>(vf, vmn);
    k_ctxfill<<<(NT * DM) / 256, 256, 0, stream>>>(vmn, abf);   // ctx (bf16)
    k_attn<<<N_B * NH * U_TOP, 256, 0, stream>>>(qf, kf, vf, topb, abf);

    // output projection + residual LN
    gemm(abf, Wo, DM, DM, bo, t0, nullptr, 0, 0);
    k_ln<<<NT, 256, 0, stream>>>(h, t0, g1, be1, h);

    // FFN: GELU fused into GEMM1 epilogue, bf16 intermediate
    cvt(h, abf, NT * DM);
    gemm(abf, W1, DFF, DM, b1, nullptr, bbf, /*gelu*/1, /*bf16 out*/1);
    gemm(bbf, W2, DM, DFF, b2, t0, nullptr, 0, 0);
    k_ln<<<NT, 256, 0, stream>>>(h, t0, g2, be2, h);
  }

  // --- final LN + projection ---
  k_ln<<<NT, 256, 0, stream>>>(h, nullptr, ln_g, ln_b, t0);
  k_proj<<<NT / 256, 256, 0, stream>>>(t0, Wp, bp, (float*)d_out);
}
